// InferenceEngine_19851338842402
// MI455X (gfx1250) — matrix operand order repaired and checked by reading, not compiled, not run on hardware
//
#include <hip/hip_runtime.h>
#include <hip/hip_bf16.h>
#include <math.h>

// ---------------------------------------------------------------------------
// InferenceEngine forward for MI455X (gfx1250, wave32, WMMA).
// Big GEMMs: v_wmma_f32_16x16x32_bf16 with double-buffered
// global_load_async_to_lds_b128 tile staging (ASYNCcnt pipeline).
// ---------------------------------------------------------------------------

typedef __attribute__((ext_vector_type(16))) __bf16 v16bf;
typedef __attribute__((ext_vector_type(8)))  float  v8f;

union FragU { uint4 q[2]; v16bf v; };

#define B_   8
#define S_   1024
#define H_   1024
#define M_   4
#define STEPS_ 5
#define ROWS (B_ * S_)          // 8192

#define TM 128
#define TN 128
#define KC 32
#define LDP 40                  // padded LDS row stride (bf16 units), 80B, 16B-aligned

__device__ __forceinline__ float gelu_exact(float x) {
    return 0.5f * x * (1.0f + erff(x * 0.70710678118654752440f));
}
__device__ __forceinline__ float sigmoidf_(float x) {
    return 1.0f / (1.0f + expf(-x));
}

// CDNA5 async copy: global -> LDS, 16B per lane, tracked by ASYNCcnt.
__device__ __forceinline__ void async_b128_to_lds(unsigned lds_off, const void* gptr) {
    asm volatile("global_load_async_to_lds_b128 %0, %1, off"
                 :
                 : "v"(lds_off), "v"((unsigned long long)(uintptr_t)gptr)
                 : "memory");
}
__device__ __forceinline__ void wait_asynccnt0() {
    asm volatile("s_wait_asynccnt 0x0" ::: "memory");
}

// ---------------------------------------------------------------------------
// WMMA GEMM:  out[M,N] = epi(A[M,K] (bf16) @ Wt[N,K]^T (bf16) + bias[N])
// Epilogue 0: GELU -> bf16 out.  Epilogue 1: bias only -> f32 out.
// M,N multiples of 128; K multiple of 32 (true for all call sites).
// ---------------------------------------------------------------------------
template <int EPI>
__global__ __launch_bounds__(256)
void gemm_bf16_wmma(const __bf16* __restrict__ A,
                    const __bf16* __restrict__ Wt,
                    const float*  __restrict__ bias,
                    void* __restrict__ out,
                    int M, int N, int K)
{
    __shared__ __attribute__((aligned(16))) __bf16 lsA[2][TM * LDP];
    __shared__ __attribute__((aligned(16))) __bf16 lsB[2][TN * LDP];

    const int tid  = threadIdx.x;
    const int lane = tid & 31;
    const int wid  = tid >> 5;
    const int wm   = wid & 3;          // 4 waves along M  -> 4*32 = 128
    const int wn   = wid >> 2;         // 2 waves along N  -> 2*64 = 128
    const int bm   = blockIdx.x * TM;
    const int bn   = blockIdx.y * TN;

    const int laneM  = lane & 15;
    const int laneHi = lane >> 4;
    const int kbA = laneHi * 8;        // A frag: elems 0..7 -> K=kbA..+7, 8..15 -> K=kbA+16..+23
    const int ksB = laneHi * 8;        // B frag: elems 0..7 -> K=ksB..+7, 8..15 -> K=ksB+16..+23

    v8f acc[2][4];
#pragma unroll
    for (int mi = 0; mi < 2; ++mi)
#pragma unroll
        for (int ni = 0; ni < 4; ++ni)
#pragma unroll
            for (int j = 0; j < 8; ++j) acc[mi][ni][j] = 0.0f;

    // Tile-staging assignment: thread -> (row 0..127, 16-elem segment 0/1)
    const int lrow = tid >> 1;           // 0..127
    const int lseg = (tid & 1) * 16;     // 0 or 16 (bf16 elems)

    // Per-thread LDS byte offsets (generic shared ptr low 32 bits == LDS addr).
    const unsigned ldsA[2] = {
        (unsigned)(uintptr_t)&lsA[0][lrow * LDP + lseg],
        (unsigned)(uintptr_t)&lsA[1][lrow * LDP + lseg] };
    const unsigned ldsB[2] = {
        (unsigned)(uintptr_t)&lsB[0][lrow * LDP + lseg],
        (unsigned)(uintptr_t)&lsB[1][lrow * LDP + lseg] };

    const __bf16* gA = A  + (size_t)(bm + lrow) * K + lseg;
    const __bf16* gB = Wt + (size_t)(bn + lrow) * K + lseg;

    auto issue_tiles = [&](int k0, int buf) {
        async_b128_to_lds(ldsA[buf],      gA + k0);
        async_b128_to_lds(ldsA[buf] + 16, gA + k0 + 8);
        async_b128_to_lds(ldsB[buf],      gB + k0);
        async_b128_to_lds(ldsB[buf] + 16, gB + k0 + 8);
    };

    // Prologue: stage chunk 0 into buffer 0.
    issue_tiles(0, 0);
    wait_asynccnt0();
    __syncthreads();

    int buf = 0;
    for (int k0 = 0; k0 < K; k0 += KC) {
        // Prefetch next chunk into the other buffer while we compute.
        if (k0 + KC < K) issue_tiles(k0 + KC, buf ^ 1);

        const __bf16* sA = lsA[buf];
        const __bf16* sB = lsB[buf];

        v16bf bfrag[4];
#pragma unroll
        for (int ni = 0; ni < 4; ++ni) {
            const __bf16* bp = &sB[(wn * 64 + ni * 16 + laneM) * LDP + ksB];
            FragU u;
            u.q[0] = *(const uint4*)bp;
            u.q[1] = *(const uint4*)(bp + 16);
            bfrag[ni] = u.v;
        }
#pragma unroll
        for (int mi = 0; mi < 2; ++mi) {
            const __bf16* ap = &sA[(wm * 32 + mi * 16 + laneM) * LDP + kbA];
            FragU u;
            u.q[0] = *(const uint4*)ap;
            u.q[1] = *(const uint4*)(ap + 16);
            v16bf afrag = u.v;
#pragma unroll
            for (int ni = 0; ni < 4; ++ni) {
                acc[mi][ni] = __builtin_amdgcn_wmma_f32_16x16x32_bf16(
                    false, afrag, false, bfrag[ni],
                    (short)0, acc[mi][ni], false, false);
            }
        }

        wait_asynccnt0();      // next chunk fully landed in LDS (this wave's ops)
        __syncthreads();       // all waves done reading buf & all async data visible
        buf ^= 1;
    }

    // Epilogue.  C/D layout: VGPR j -> row j (lanes 0-15) / j+8 (lanes 16-31), col = lane&15.
    const int rsub = laneHi * 8;
#pragma unroll
    for (int mi = 0; mi < 2; ++mi) {
#pragma unroll
        for (int ni = 0; ni < 4; ++ni) {
            const int col = bn + wn * 64 + ni * 16 + laneM;
            const float bv = bias[col];
#pragma unroll
            for (int j = 0; j < 8; ++j) {
                const int row = bm + wm * 32 + mi * 16 + rsub + j;
                float v = acc[mi][ni][j] + bv;
                if (EPI == 0) {
                    v = gelu_exact(v);
                    ((__bf16*)out)[(size_t)row * N + col] = (__bf16)v;
                } else {
                    ((float*)out)[(size_t)row * N + col] = v;
                }
            }
        }
    }
}

// ---------------------------------------------------------------------------
// Elementwise / reduction helpers
// ---------------------------------------------------------------------------
__global__ void k_fill_zero(float* __restrict__ p, size_t n) {
    size_t i = (size_t)blockIdx.x * blockDim.x + threadIdx.x;
    if (i < n) p[i] = 0.0f;
}

__global__ void k_cvt_bf16(const float* __restrict__ in, __bf16* __restrict__ out, size_t n) {
    size_t i = (size_t)blockIdx.x * blockDim.x + threadIdx.x;
    if (i < n) out[i] = (__bf16)in[i];
}

// W[K,N] f32 -> Wt[N,K] bf16
__global__ void k_transpose_cvt(const float* __restrict__ W, __bf16* __restrict__ Wt,
                                int K, int N) {
    size_t i = (size_t)blockIdx.x * blockDim.x + threadIdx.x;
    if (i >= (size_t)K * N) return;
    int n = (int)(i % N);
    int k = (int)(i / N);
    Wt[(size_t)n * K + k] = (__bf16)W[i];
}

// combined[r, 0:1024] = state_bf[r], combined[r, 1024:2048] = ctx_bf[r/1024]
__global__ void k_pack_combined(const __bf16* __restrict__ state_bf,
                                const __bf16* __restrict__ ctx_bf,
                                __bf16* __restrict__ comb) {
    size_t i = (size_t)blockIdx.x * blockDim.x + threadIdx.x;
    if (i >= (size_t)ROWS * 2 * H_) return;
    int c = (int)(i & (2 * H_ - 1));
    size_t r = i >> 11;
    comb[i] = (c < H_) ? state_bf[(r << 10) + c]
                       : ctx_bf[((r >> 10) << 10) + (c - H_)];
}

// Row LayerNorm: out = (x-mu)*rsqrt(var+eps)*g + b ; optional bf16 mirror.
__global__ __launch_bounds__(256)
void k_layernorm(const float* __restrict__ in, const float* __restrict__ g,
                 const float* __restrict__ b, float* __restrict__ out_f32,
                 __bf16* __restrict__ out_bf16, int N) {
    const int row = blockIdx.x;
    const int tid = threadIdx.x;
    const float* x = in + (size_t)row * N;
    float s = 0.0f, s2 = 0.0f;
    for (int i = tid; i < N; i += 256) { float v = x[i]; s += v; s2 += v * v; }
    __shared__ float sh_s[256], sh_q[256];
    sh_s[tid] = s; sh_q[tid] = s2;
    __syncthreads();
    for (int off = 128; off > 0; off >>= 1) {
        if (tid < off) { sh_s[tid] += sh_s[tid + off]; sh_q[tid] += sh_q[tid + off]; }
        __syncthreads();
    }
    const float mean = sh_s[0] / (float)N;
    const float var  = sh_q[0] / (float)N - mean * mean;
    const float inv  = rsqrtf(var + 1e-5f);
    for (int i = tid; i < N; i += 256) {
        float v = (x[i] - mean) * inv * g[i] + b[i];
        if (out_f32)  out_f32[(size_t)row * N + i] = v;
        if (out_bf16) out_bf16[(size_t)row * N + i] = (__bf16)v;
    }
}

// GRU gate fusion + gated state + depth-weighted accumulation of explicit_result.
__global__ void k_gru_combine(const float* __restrict__ gi, const float* __restrict__ gh,
                              const float* __restrict__ nstate,
                              float* __restrict__ mem_f, __bf16* __restrict__ mem_bf,
                              __bf16* __restrict__ state_bf,
                              const float* __restrict__ depth_w, int step,
                              float* __restrict__ expl_acc) {
    size_t idx = (size_t)blockIdx.x * blockDim.x + threadIdx.x;
    if (idx >= (size_t)ROWS * H_) return;
    const size_t r = idx >> 10;
    const int    j = (int)(idx & (H_ - 1));
    const int    bb = (int)(r >> 10);
    const size_t gb = r * (size_t)(3 * H_) + j;
    float ir = gi[gb], iz = gi[gb + H_], in_ = gi[gb + 2 * H_];
    float hr = gh[gb], hz = gh[gb + H_], hn  = gh[gb + 2 * H_];
    float rg = sigmoidf_(ir + hr);
    float zg = sigmoidf_(iz + hz);
    float ng = tanhf(in_ + rg * hn);
    float hprev = mem_f[idx];
    float m  = (1.0f - zg) * ng + zg * hprev;
    float gated = nstate[idx] + 0.1f * m;
    mem_f[idx]   = m;
    mem_bf[idx]  = (__bf16)m;
    state_bf[idx] = (__bf16)gated;
    expl_acc[idx] += depth_w[bb * STEPS_ + step] * gated;
}

// context heads (tiny): hidden = gelu(ctx @ W + b)
__global__ void k_head_hidden(const float* __restrict__ ctx, const float* __restrict__ W,
                              const float* __restrict__ bias, float* __restrict__ out,
                              int K, int N) {
    int idx = blockIdx.x * blockDim.x + threadIdx.x;
    if (idx >= B_ * N) return;
    int bb = idx / N, n = idx % N;
    const float* x = ctx + (size_t)bb * K;
    float s = bias[n];
    for (int k = 0; k < K; ++k) s += x[k] * W[(size_t)k * N + n];
    out[idx] = gelu_exact(s);
}

// out[8,C] = softmax(hidden @ W + b, axis=-1); single block.
__global__ void k_head_out_softmax(const float* __restrict__ hid, const float* __restrict__ W,
                                   const float* __restrict__ bias, float* __restrict__ out,
                                   int K, int C) {
    __shared__ float logits[B_][8];
    int t = threadIdx.x;
    if (t < B_ * C) {
        int bb = t / C, c = t % C;
        const float* x = hid + (size_t)bb * K;
        float s = bias[c];
        for (int k = 0; k < K; ++k) s += x[k] * W[(size_t)k * C + c];
        logits[bb][c] = s;
    }
    __syncthreads();
    if (t < B_) {
        float mx = -1e30f;
        for (int c = 0; c < C; ++c) mx = fmaxf(mx, logits[t][c]);
        float den = 0.0f, e[8];
        for (int c = 0; c < C; ++c) { e[c] = expf(logits[t][c] - mx); den += e[c]; }
        for (int c = 0; c < C; ++c) out[t * C + c] = e[c] / den;
    }
}

// implicit_det[r,h] = mean over M of implicit[r, m*H + h]
__global__ void k_impl_det(const float* __restrict__ impl, float* __restrict__ det) {
    size_t idx = (size_t)blockIdx.x * blockDim.x + threadIdx.x;
    if (idx >= (size_t)ROWS * H_) return;
    size_t r = idx >> 10;
    int h = (int)(idx & (H_ - 1));
    const float* p = impl + r * (size_t)(M_ * H_) + h;
    det[idx] = 0.25f * (p[0] + p[H_] + p[2 * H_] + p[3 * H_]);
}

// result = mw[b,0]*det + mw[b,1]*expl
__global__ void k_final_combine(const float* __restrict__ det, const float* __restrict__ expl,
                                const float* __restrict__ mw, float* __restrict__ res) {
    size_t idx = (size_t)blockIdx.x * blockDim.x + threadIdx.x;
    if (idx >= (size_t)ROWS * H_) return;
    int bb = (int)(idx >> 20);
    res[idx] = mw[bb * 2 + 0] * det[idx] + mw[bb * 2 + 1] * expl[idx];
}

// ---------------------------------------------------------------------------
// Host orchestration
// ---------------------------------------------------------------------------
static inline unsigned gblocks(size_t n) { return (unsigned)((n + 255) / 256); }

extern "C" void kernel_launch(void* const* d_in, const int* in_sizes, int n_in,
                              void* d_out, int out_size, void* d_ws, size_t ws_size,
                              hipStream_t stream) {
    (void)in_sizes; (void)n_in; (void)out_size; (void)ws_size;

    const float* det_state = (const float*)d_in[0];   // [8,1024,1024]
    const float* sup_state = (const float*)d_in[1];   // [8,1024,4096]
    const float* context   = (const float*)d_in[2];   // [8,1024]
    const float* Wm1 = (const float*)d_in[3];  const float* bm1 = (const float*)d_in[4];
    const float* Wm2 = (const float*)d_in[5];  const float* bm2 = (const float*)d_in[6];
    const float* Wd1 = (const float*)d_in[7];  const float* bd1 = (const float*)d_in[8];
    const float* Wd2 = (const float*)d_in[9];  const float* bd2 = (const float*)d_in[10];
    const float* We1 = (const float*)d_in[11]; const float* be1 = (const float*)d_in[12];
    const float* We2 = (const float*)d_in[13]; const float* be2 = (const float*)d_in[14];
    const float* ln_g = (const float*)d_in[15]; const float* ln_b = (const float*)d_in[16];
    const float* W_ih = (const float*)d_in[17]; const float* W_hh = (const float*)d_in[18];
    const float* b_ih = (const float*)d_in[19]; const float* b_hh = (const float*)d_in[20];
    const float* Wi1 = (const float*)d_in[21]; const float* bi1 = (const float*)d_in[22];
    const float* Wi2 = (const float*)d_in[23]; const float* bi2 = (const float*)d_in[24];
    const float* lni_g = (const float*)d_in[25]; const float* lni_b = (const float*)d_in[26];

    float* out = (float*)d_out;
    float* out_result = out;                          // 8388608
    float* out_mode   = out + 8388608;                // 16
    float* out_depth  = out + 8388624;                // 40
    float* out_expl   = out + 8388664;                // 8388608
    float* out_impl   = out + 16777272;               // 33554432

    // ---- workspace layout (bump allocator; all sizes 256B multiples) ----
    char* ws = (char*)d_ws;
    size_t o = 0;
    auto alloc = [&](size_t bytes) { size_t r = o; o += (bytes + 255) & ~(size_t)255; return r; };

    __bf16* we1t = (__bf16*)(ws + alloc((size_t)STEPS_ * 2 * H_ * H_ * 2));  // [5][1024][2048]
    __bf16* we2t = (__bf16*)(ws + alloc((size_t)STEPS_ * H_ * H_ * 2));      // [5][1024][1024]
    __bf16* wiht = (__bf16*)(ws + alloc((size_t)3 * H_ * H_ * 2));           // [3072][1024]
    __bf16* whht = (__bf16*)(ws + alloc((size_t)3 * H_ * H_ * 2));
    __bf16* wi1t = (__bf16*)(ws + alloc((size_t)(M_ * H_) * 2 * H_ * 2));    // [2048][4096]
    __bf16* wi2t = (__bf16*)(ws + alloc((size_t)2 * H_ * (M_ * H_) * 2));    // [4096][2048]
    __bf16* state_bf = (__bf16*)(ws + alloc((size_t)ROWS * H_ * 2));
    __bf16* ctx_bf   = (__bf16*)(ws + alloc((size_t)B_ * H_ * 2));
    float*  hidM = (float*)(ws + alloc((size_t)B_ * H_ * 4));
    float*  hidD = (float*)(ws + alloc((size_t)B_ * (H_ / 2) * 4));
    float*  mem_f  = (float*)(ws + alloc((size_t)ROWS * H_ * 4));
    __bf16* mem_bf = (__bf16*)(ws + alloc((size_t)ROWS * H_ * 2));
    char*   scratch = ws + alloc((size_t)335544320);

    // explicit-loop view of scratch
    __bf16* comb_bf   = (__bf16*)(scratch);                          // 33554432 B
    __bf16* h1_bf     = (__bf16*)(scratch + 33554432);               // 16777216 B
    float*  preln     = (float*)(scratch + 50331648);                // 33554432 B
    float*  nstate_f  = (float*)(scratch + 83886080);                // 33554432 B
    __bf16* nstate_bf = (__bf16*)(scratch + 117440512);              // 16777216 B
    float*  gi        = (float*)(scratch + 134217728);               // 100663296 B
    float*  gh        = (float*)(scratch + 234881024);               // 100663296 B
    // implicit-phase view of same scratch (runs after the loop)
    __bf16* sup_bf   = (__bf16*)(scratch);                           // 67108864 B
    __bf16* hi_bf    = (__bf16*)(scratch + 67108864);                // 33554432 B
    float*  pre_impl = (float*)(scratch + 100663296);                // 134217728 B
    float*  impl_det = (float*)(scratch + 234881024);                // 33554432 B

    const size_t nRH  = (size_t)ROWS * H_;           // 8388608
    const size_t nR2H = (size_t)ROWS * 2 * H_;       // 16777216
    const size_t nSUP = (size_t)ROWS * M_ * H_;      // 33554432

    // ---- init: zero accumulators / GRU memory ----
    k_fill_zero<<<gblocks(nRH), 256, 0, stream>>>(out_expl, nRH);
    k_fill_zero<<<gblocks(nRH), 256, 0, stream>>>(mem_f, nRH);
    k_fill_zero<<<gblocks(nRH / 2), 256, 0, stream>>>((float*)mem_bf, nRH / 2);

    // ---- weight conversion (f32 [K,N] -> bf16 [N,K]) ----
    for (int s = 0; s < STEPS_; ++s) {
        k_transpose_cvt<<<gblocks((size_t)2 * H_ * H_), 256, 0, stream>>>(
            We1 + (size_t)s * 2 * H_ * H_, we1t + (size_t)s * 2 * H_ * H_, 2 * H_, H_);
        k_transpose_cvt<<<gblocks((size_t)H_ * H_), 256, 0, stream>>>(
            We2 + (size_t)s * H_ * H_, we2t + (size_t)s * H_ * H_, H_, H_);
    }
    k_transpose_cvt<<<gblocks((size_t)H_ * 3 * H_), 256, 0, stream>>>(W_ih, wiht, H_, 3 * H_);
    k_transpose_cvt<<<gblocks((size_t)H_ * 3 * H_), 256, 0, stream>>>(W_hh, whht, H_, 3 * H_);
    k_transpose_cvt<<<gblocks((size_t)M_ * H_ * 2 * H_), 256, 0, stream>>>(Wi1, wi1t, M_ * H_, 2 * H_);
    k_transpose_cvt<<<gblocks((size_t)2 * H_ * M_ * H_), 256, 0, stream>>>(Wi2, wi2t, 2 * H_, M_ * H_);

    // ---- activation conversion ----
    k_cvt_bf16<<<gblocks(nRH), 256, 0, stream>>>(det_state, state_bf, nRH);
    k_cvt_bf16<<<gblocks((size_t)B_ * H_), 256, 0, stream>>>(context, ctx_bf, (size_t)B_ * H_);

    // ---- context heads (mode / depth weights) ----
    k_head_hidden<<<gblocks((size_t)B_ * H_), 256, 0, stream>>>(context, Wm1, bm1, hidM, H_, H_);
    k_head_out_softmax<<<1, 256, 0, stream>>>(hidM, Wm2, bm2, out_mode, H_, 2);
    k_head_hidden<<<gblocks((size_t)B_ * (H_ / 2)), 256, 0, stream>>>(context, Wd1, bd1, hidD, H_, H_ / 2);
    k_head_out_softmax<<<1, 256, 0, stream>>>(hidD, Wd2, bd2, out_depth, H_ / 2, STEPS_);

    // ---- explicit reasoning loop ----
    for (int s = 0; s < STEPS_; ++s) {
        k_pack_combined<<<gblocks(nR2H), 256, 0, stream>>>(state_bf, ctx_bf, comb_bf);

        gemm_bf16_wmma<0><<<dim3(ROWS / TM, H_ / TN), 256, 0, stream>>>(
            comb_bf, we1t + (size_t)s * 2 * H_ * H_, be1 + (size_t)s * H_,
            h1_bf, ROWS, H_, 2 * H_);

        gemm_bf16_wmma<1><<<dim3(ROWS / TM, H_ / TN), 256, 0, stream>>>(
            h1_bf, we2t + (size_t)s * H_ * H_, be2 + (size_t)s * H_,
            preln, ROWS, H_, H_);

        k_layernorm<<<ROWS, 256, 0, stream>>>(preln, ln_g + (size_t)s * H_, ln_b + (size_t)s * H_,
                                              nstate_f, nstate_bf, H_);

        gemm_bf16_wmma<1><<<dim3(ROWS / TM, (3 * H_) / TN), 256, 0, stream>>>(
            nstate_bf, wiht, b_ih, gi, ROWS, 3 * H_, H_);
        gemm_bf16_wmma<1><<<dim3(ROWS / TM, (3 * H_) / TN), 256, 0, stream>>>(
            mem_bf, whht, b_hh, gh, ROWS, 3 * H_, H_);

        k_gru_combine<<<gblocks(nRH), 256, 0, stream>>>(gi, gh, nstate_f, mem_f, mem_bf,
                                                        state_bf, out_depth, s, out_expl);
    }

    // ---- implicit path ----
    k_cvt_bf16<<<gblocks(nSUP), 256, 0, stream>>>(sup_state, sup_bf, nSUP);

    gemm_bf16_wmma<0><<<dim3(ROWS / TM, (2 * H_) / TN), 256, 0, stream>>>(
        sup_bf, wi1t, bi1, hi_bf, ROWS, 2 * H_, M_ * H_);

    gemm_bf16_wmma<1><<<dim3(ROWS / TM, (M_ * H_) / TN), 256, 0, stream>>>(
        hi_bf, wi2t, bi2, pre_impl, ROWS, M_ * H_, 2 * H_);

    k_layernorm<<<ROWS, 256, 0, stream>>>(pre_impl, lni_g, lni_b, out_impl, (__bf16*)nullptr, M_ * H_);

    k_impl_det<<<gblocks(nRH), 256, 0, stream>>>(out_impl, impl_det);

    // ---- final mix ----
    k_final_combine<<<gblocks(nRH), 256, 0, stream>>>(impl_det, out_expl, out_mode, out_result);
}
